// PharmRecGVP_17102559773369
// MI455X (gfx1250) — matrix operand order repaired and checked by reading, not compiled, not run on hardware
//
#include <hip/hip_runtime.h>
#include <hip/hip_bf16.h>
#include <math.h>

// ---------------------------------------------------------------------------
// PharmRecGVP on MI455X (gfx1250).
// - All GVP matmuls on v_wmma_f32_16x16x32_bf16 (f32 accumulate).
// - A operands in LDS as bf16, zero-padded to K%32==0 => unconditional
//   ds_load_b128 fragment loads (addrspace(3) inferred after forced inlining).
// - B operands pre-packed per launch into CDNA5 B-fragment lane layout in one
//   pool passed as a top-level kernel arg => global_load_b128.
// - s_out matmul D-stores write directly into the next GVP's A operand (safe:
//   all A fragments are preloaded to VGPRs before any D store; same-wave DS
//   ops are in order) => no intermediate SO buffer / copy loop.
// - Bias/activation resolved at compile time; wave32 per block (EXEC all-1s).
// - segment_sum via global_atomic_add_f32 (L2 atomic units).
// ---------------------------------------------------------------------------

typedef __attribute__((ext_vector_type(16))) __bf16 v16bf;
typedef __attribute__((ext_vector_type(8)))  __bf16 v8bf;
typedef __attribute__((ext_vector_type(8)))  float  v8f;

#define XLD 288   // max padded K of scalar matmul (update GVP0: 256+32)
#define VLD 32    // padded vector-channel count (and K of vector matmuls)

struct PackedOff {      // offsets into the packed fragment pool (v16bf units)
  int W, Wg, Wh, Wu;
};

__device__ __forceinline__ v8f zero8() {
  v8f z;
#pragma unroll
  for (int i = 0; i < 8; ++i) z[i] = 0.f;
  return z;
}

// A fragment 16x32 (MxK) from bf16 LDS, zero-padded source, no masking.
// lanes 0-15: M=lane, K pairs {0..7} then {16..23}; lanes 16-31: K offset +8.
__device__ __forceinline__ v16bf load_A_bf(const __bf16* src, int ld, int k0) {
  const int lane = threadIdx.x & 31;
  const int base = (lane & 15) * ld + k0 + ((lane >> 4) << 3);
  const v8bf lo = *(const v8bf*)(src + base);        // ds_load_b128
  const v8bf hi = *(const v8bf*)(src + base + 16);   // ds_load_b128
  v16bf a;
#pragma unroll
  for (int i = 0; i < 8; ++i) { a[i] = lo[i]; a[8 + i] = hi[i]; }
  return a;
}

__device__ __forceinline__ v8f wmma_bf16(v16bf a, v16bf b, v8f c) {
  return __builtin_amdgcn_wmma_f32_16x16x32_bf16(false, a, false, b,
                                                 (short)0, c, false, false);
}

// D store: VGPR r -> M = r + 8*(lane>>4), N = (lane&15)+n0.
// ACT: 0 none, 1 silu, 2 sigmoid.  HASBIAS resolved at compile time.
template <int NCOLS, int NBIAS, int ACT, bool HASBIAS, typename OutT>
__device__ __forceinline__ void store_D(OutT* out, int ld, int n0, v8f acc,
                                        const float* bias) {
  const int lane = threadIdx.x & 31;
  const int n    = (lane & 15) + n0;
  const int mb   = (lane >> 4) << 3;
  if (n < NCOLS) {
    float bv = 0.f;
    if (HASBIAS) bv = (n < NBIAS) ? bias[n] : 0.f;
#pragma unroll
    for (int r = 0; r < 8; ++r) {
      float x = HASBIAS ? (acc[r] + bv) : acc[r];
      if (ACT == 1)      x = x * (1.f / (1.f + __expf(-x)));  // silu
      else if (ACT == 2) x = 1.f / (1.f + __expf(-x));        // sigmoid
      out[(mb + r) * ld + n] = (OutT)x;
    }
  }
}

// out(16 x NCOLS) = ACT( X(16 x KPAD) @ Bp + bias ); X bf16 LDS zero-padded,
// Bp packed fragments [ntile][ktile][lane] of v16bf in the global pool.
// NOTE: all A fragments are preloaded to VGPRs before the first D store, so
// `out` may alias `X` (in-place s_out -> next A operand).
template <int KPAD, int NCOLS, int NBIAS, int ACT, bool HASBIAS, typename OutT>
__device__ __forceinline__ void mm_tile_p(const __bf16* X, int xld,
                                          const v16bf* Bp, const float* bias,
                                          OutT* out, int old) {
  constexpr int KT = KPAD / 32;
  constexpr int NT = (NCOLS + 15) / 16;
  const int lane = threadIdx.x & 31;
  __builtin_prefetch(Bp, 0, 0);  // global_prefetch_b8 (weights L2-resident)
  v16bf afr[KT];
#pragma unroll
  for (int kt = 0; kt < KT; ++kt) afr[kt] = load_A_bf(X, xld, kt * 32);
#pragma unroll
  for (int nt = 0; nt < NT; ++nt) {
    v8f acc = zero8();
#pragma unroll
    for (int kt = 0; kt < KT; ++kt) {
      const v16bf b = Bp[(nt * KT + kt) * 32 + lane];  // 2x global_load_b128
      acc = wmma_bf16(afr[kt], b, acc);
    }
    store_D<NCOLS, NBIAS, ACT, HASBIAS, OutT>(out, old, nt * 16, acc, bias);
  }
}

// One GVP on a 16-row tile.
// Entry: Xs cols [0,SIN) = s (bf16), Vb channels [0,VIN) = v, [VIN,32)=0.
// Exit:  Xs cols [0,SOUT) = s_out, Vb channels [0,VOUT) gated vu, rest 0.
template <int VIN, int VOUT, int SIN, int SOUT, int H, bool VSIG>
__device__ __forceinline__ void gvp_tile(const v16bf* __restrict__ pk,
                                         PackedOff o, const float* __restrict__ b,
                                         const float* __restrict__ bg,
                                         __bf16* Xs, __bf16* Vb, __bf16* VH,
                                         float* VU, float* GT) {
  constexpr int KPAD = ((SIN + H + 31) / 32) * 32;
  const int lane = threadIdx.x;

  // 1) vh_c = v_c @ Wh : (16x32)(32x32); Wh N-padded to 32 => VH zero >= H.
#pragma unroll
  for (int c = 0; c < 3; ++c)
    mm_tile_p<32, 32, 0, 0, false, __bf16>(Vb + c * 16 * VLD, VLD, pk + o.Wh,
                                           nullptr, VH + c * 16 * VLD, VLD);
  __syncthreads();

  // 2) vu_c = vh_c @ Wu : (16x32)(32x16)
#pragma unroll
  for (int c = 0; c < 3; ++c)
    mm_tile_p<32, 16, 0, 0, false, float>(VH + c * 16 * VLD, VLD, pk + o.Wu,
                                          nullptr, VU + c * 256, 16);
  __syncthreads();

  // 3) sh = ||vh|| appended at Xs[SIN, KPAD); zero beyond H since VH pad = 0.
  constexpr int PADW = KPAD - SIN;
  for (int idx = lane; idx < 16 * PADW; idx += 32) {
    const int r = idx / PADW, j = idx - r * PADW;
    const float a0 = (float)VH[0 * 16 * VLD + r * VLD + j];
    const float a1 = (float)VH[1 * 16 * VLD + r * VLD + j];
    const float a2 = (float)VH[2 * 16 * VLD + r * VLD + j];
    Xs[r * XLD + SIN + j] = (__bf16)sqrtf(a0 * a0 + a1 * a1 + a2 * a2);
  }
  __syncthreads();

  // 4) s_out = silu([s,sh] @ W + b) -- D-stores land directly in Xs[0,SOUT)
  //    (A fragments preloaded before stores; same-wave DS ops in order).
  mm_tile_p<KPAD, SOUT, SOUT, 1, true, __bf16>(Xs, XLD, pk + o.W, b, Xs, XLD);
  __syncthreads();

  // 5) gate = (sigmoid?)(s_out @ Wg + bg), A re-read from Xs[0,SOUT)
  mm_tile_p<SOUT, 16, VOUT, VSIG ? 2 : 0, true, float>(Xs, XLD, pk + o.Wg, bg,
                                                       GT, 16);
  __syncthreads();

  // 6) gated vector output (Vb re-zero-padded beyond VOUT)
  for (int idx = lane; idx < 16 * 3 * 32; idx += 32) {
    const int r = idx / 96, rem = idx - r * 96;
    const int c = rem >> 5, u = rem & 31;
    Vb[c * 16 * VLD + r * VLD + u] =
        (u < VOUT) ? (__bf16)(GT[r * 16 + u] * VU[c * 256 + r * 16 + u])
                   : (__bf16)0.f;
  }
  __syncthreads();
}

#define GVP_LDS                                                                \
  __shared__ __align__(16) __bf16 Xs[16 * XLD];                                \
  __shared__ __align__(16) __bf16 Vb[3 * 16 * VLD];                            \
  __shared__ __align__(16) __bf16 VH[3 * 16 * VLD];                            \
  __shared__ float VU[3 * 16 * 16];                                            \
  __shared__ float GT[16 * 16];

// ------------------------- message kernel (per edge type) ------------------
__global__ __launch_bounds__(32)
void gvp_msg_kernel(const float* __restrict__ s_src, const float* __restrict__ v_src,
                    const float* __restrict__ x_src, const float* __restrict__ x_dst,
                    const int* __restrict__ e_src, const int* __restrict__ e_dst,
                    int n_edges, const v16bf* __restrict__ pk,
                    PackedOff o0, PackedOff o1, PackedOff o2,
                    const float* __restrict__ b0, const float* __restrict__ bg0,
                    const float* __restrict__ b1, const float* __restrict__ bg1,
                    const float* __restrict__ b2, const float* __restrict__ bg2,
                    float* __restrict__ agg_s, float* __restrict__ agg_v) {
  GVP_LDS
  __shared__ int sI[16], dI[16];
  const int lane = threadIdx.x;
  const int e0   = blockIdx.x * 16;

  if (lane < 16) {
    int e = e0 + lane;
    if (e >= n_edges) e = n_edges - 1;
    sI[lane] = e_src[e];
    dI[lane] = e_dst[e];
  }
  __syncthreads();

  // scalars: cols [0,128)=s0[si], col 128 = d  (SIN=129)
  for (int idx = lane; idx < 16 * 128; idx += 32) {
    const int r = idx >> 7, cc = idx & 127;
    Xs[r * XLD + cc] = (__bf16)s_src[sI[r] * 128 + cc];
  }
  // vectors: channel 0 = unit(xdiff), channels 1..16 = v0[si], 17..31 = 0
  for (int idx = lane; idx < 16 * 48; idx += 32) {
    const int r = idx / 48, rem = idx - r * 48, u = rem / 3, c = rem - u * 3;
    Vb[c * 16 * VLD + r * VLD + (u + 1)] = (__bf16)v_src[(sI[r] * 16 + u) * 3 + c];
  }
  for (int idx = lane; idx < 16 * 3 * 15; idx += 32) {  // zero pad [17,32)
    const int r = idx / 45, rem = idx - r * 45, c = rem / 15, u = rem - c * 15;
    Vb[c * 16 * VLD + r * VLD + 17 + u] = (__bf16)0.f;
  }
  if (lane < 16) {
    const int r = lane;
    const float dx0 = x_dst[dI[r] * 3 + 0] - x_src[sI[r] * 3 + 0];
    const float dx1 = x_dst[dI[r] * 3 + 1] - x_src[sI[r] * 3 + 1];
    const float dx2 = x_dst[dI[r] * 3 + 2] - x_src[sI[r] * 3 + 2];
    const float d   = sqrtf(dx0 * dx0 + dx1 * dx1 + dx2 * dx2);
    Xs[r * XLD + 128] = (__bf16)d;
    const float inv = 1.f / (d + 1e-8f);
    Vb[0 * 16 * VLD + r * VLD] = (__bf16)(dx0 * inv);
    Vb[1 * 16 * VLD + r * VLD] = (__bf16)(dx1 * inv);
    Vb[2 * 16 * VLD + r * VLD] = (__bf16)(dx2 * inv);
  }
  __syncthreads();

  gvp_tile<17, 16, 129, 128, 17, true>(pk, o0, b0, bg0, Xs, Vb, VH, VU, GT);
  gvp_tile<16, 16, 128, 128, 16, true>(pk, o1, b1, bg1, Xs, Vb, VH, VU, GT);
  gvp_tile<16, 16, 128, 128, 16, true>(pk, o2, b2, bg2, Xs, Vb, VH, VU, GT);

  // segment-sum scatter (scaled by 1/MSG_NORM) via f32 L2 atomics
  const float invn = 0.1f;
  for (int idx = lane; idx < 16 * 128; idx += 32) {
    const int r = idx >> 7, cc = idx & 127;
    if (e0 + r < n_edges)
      __hip_atomic_fetch_add(&agg_s[dI[r] * 128 + cc],
                             (float)Xs[r * XLD + cc] * invn,
                             __ATOMIC_RELAXED, __HIP_MEMORY_SCOPE_AGENT);
  }
  for (int idx = lane; idx < 16 * 48; idx += 32) {
    const int r = idx / 48, rem = idx - r * 48, u = rem / 3, c = rem - u * 3;
    if (e0 + r < n_edges)
      __hip_atomic_fetch_add(&agg_v[(dI[r] * 16 + u) * 3 + c],
                             (float)Vb[c * 16 * VLD + r * VLD + u] * invn,
                             __ATOMIC_RELAXED, __HIP_MEMORY_SCOPE_AGENT);
  }
}

// ------------------------- node update kernel ------------------------------
__global__ __launch_bounds__(32)
void gvp_upd_kernel(float* __restrict__ s_node, float* __restrict__ v_node,
                    const float* __restrict__ agg_s, const float* __restrict__ agg_v,
                    int n_nodes, const v16bf* __restrict__ pk,
                    PackedOff o0, PackedOff o1,
                    const float* __restrict__ b0, const float* __restrict__ bg0,
                    const float* __restrict__ b1, const float* __restrict__ bg1) {
  GVP_LDS
  const int lane = threadIdx.x;
  const int n0   = blockIdx.x * 16;

  for (int idx = lane; idx < 16 * 128; idx += 32) {
    const int r = idx >> 7, cc = idx & 127;
    int node = n0 + r; if (node >= n_nodes) node = n_nodes - 1;
    Xs[r * XLD + cc]       = (__bf16)s_node[node * 128 + cc];
    Xs[r * XLD + 128 + cc] = (__bf16)agg_s[node * 128 + cc];
  }
  for (int idx = lane; idx < 16 * 48; idx += 32) {
    const int r = idx / 48, rem = idx - r * 48, u = rem / 3, c = rem - u * 3;
    int node = n0 + r; if (node >= n_nodes) node = n_nodes - 1;
    Vb[c * 16 * VLD + r * VLD + u]      = (__bf16)v_node[(node * 16 + u) * 3 + c];
    Vb[c * 16 * VLD + r * VLD + 16 + u] = (__bf16)agg_v[(node * 16 + u) * 3 + c];
  }
  __syncthreads();

  gvp_tile<32, 16, 256, 128, 32, true>(pk, o0, b0, bg0, Xs, Vb, VH, VU, GT);
  gvp_tile<16, 16, 128, 128, 16, true>(pk, o1, b1, bg1, Xs, Vb, VH, VU, GT);

  // residual, in place (each node owned by exactly one tile)
  for (int idx = lane; idx < 16 * 128; idx += 32) {
    const int r = idx >> 7, cc = idx & 127;
    const int node = n0 + r;
    if (node < n_nodes) s_node[node * 128 + cc] += (float)Xs[r * XLD + cc];
  }
  for (int idx = lane; idx < 16 * 48; idx += 32) {
    const int r = idx / 48, rem = idx - r * 48, u = rem / 3, c = rem - u * 3;
    const int node = n0 + r;
    if (node < n_nodes)
      v_node[(node * 16 + u) * 3 + c] += (float)Vb[c * 16 * VLD + r * VLD + u];
  }
}

// ------------------------- noise head --------------------------------------
__global__ __launch_bounds__(32)
void gvp_noise_kernel(const float* __restrict__ s_node, const float* __restrict__ v_node,
                      int n_nodes, const v16bf* __restrict__ pk,
                      PackedOff o0, PackedOff o1, PackedOff o2, int oWout,
                      const float* __restrict__ b0, const float* __restrict__ bg0,
                      const float* __restrict__ b1, const float* __restrict__ bg1,
                      const float* __restrict__ b2, const float* __restrict__ bg2,
                      const float* __restrict__ b_out,
                      float* __restrict__ out_s, float* __restrict__ out_v) {
  GVP_LDS
  const int lane = threadIdx.x;
  const int n0   = blockIdx.x * 16;

  for (int idx = lane; idx < 16 * 128; idx += 32) {
    const int r = idx >> 7, cc = idx & 127;
    int node = n0 + r; if (node >= n_nodes) node = n_nodes - 1;
    Xs[r * XLD + cc] = (__bf16)s_node[node * 128 + cc];
  }
  for (int idx = lane; idx < 16 * 48; idx += 32) {
    const int r = idx / 48, rem = idx - r * 48, u = rem / 3, c = rem - u * 3;
    int node = n0 + r; if (node >= n_nodes) node = n_nodes - 1;
    Vb[c * 16 * VLD + r * VLD + u] = (__bf16)v_node[(node * 16 + u) * 3 + c];
  }
  for (int idx = lane; idx < 16 * 48; idx += 32) {  // zero pad channels [16,32)
    const int r = idx / 48, rem = idx - r * 48, c = rem >> 4, u = rem & 15;
    Vb[c * 16 * VLD + r * VLD + 16 + u] = (__bf16)0.f;
  }
  __syncthreads();

  gvp_tile<16, 16, 128, 128, 16, true>(pk, o0, b0, bg0, Xs, Vb, VH, VU, GT);
  gvp_tile<16, 16, 128, 128, 16, true>(pk, o1, b1, bg1, Xs, Vb, VH, VU, GT);
  gvp_tile<16, 1, 128, 64, 16, false>(pk, o2, b2, bg2, Xs, Vb, VH, VU, GT);

  // scalar_noise = s(64) @ W_out(64x128) + b_out -> D-store straight to global
  mm_tile_p<64, 128, 128, 0, true, float>(Xs, XLD, pk + oWout, b_out,
                                          out_s + n0 * 128, 128);

  if (lane < 16) {
    const int node = n0 + lane;
    if (node < n_nodes)
      for (int c = 0; c < 3; ++c)
        out_v[node * 3 + c] = (float)Vb[c * 16 * VLD + lane * VLD + 0];
  }
}

// ------------------------- pack / fill / copy helpers -----------------------
// Pack W (K x N, row-major f32) into bf16 B-fragments, zero-padded:
// out ushort index = ((nt*KT + kt)*32 + lane)*16 + e ;
// n = nt*16 + (lane&15); kk = kt*32 + (lane>>4)*8 + (e&7) + (e>>3)*16.
__global__ void pack_B_kernel(const float* __restrict__ W, int K, int N,
                              int KT, int NT, unsigned short* __restrict__ out) {
  const int total = NT * KT * 512;
  for (int idx = blockIdx.x * blockDim.x + threadIdx.x; idx < total;
       idx += gridDim.x * blockDim.x) {
    const int e    = idx & 15;
    const int t    = idx >> 4;
    const int lane = t & 31;
    const int frag = t >> 5;
    const int kt   = frag % KT;
    const int nt   = frag / KT;
    const int n    = nt * 16 + (lane & 15);
    const int kk   = kt * 32 + ((lane >> 4) << 3) + (e & 7) + ((e >> 3) << 4);
    const float v  = (kk < K && n < N) ? W[kk * N + n] : 0.f;
    const __bf16 h = (__bf16)v;
    out[idx] = *(const unsigned short*)&h;
  }
}
__global__ void fill_kernel(float* __restrict__ p, float v, int n) {
  for (int i = blockIdx.x * blockDim.x + threadIdx.x; i < n;
       i += gridDim.x * blockDim.x) p[i] = v;
}
__global__ void copy_kernel(float* __restrict__ dst, const float* __restrict__ src,
                            int n) {
  for (int i = blockIdx.x * blockDim.x + threadIdx.x; i < n;
       i += gridDim.x * blockDim.x) dst[i] = src[i];
}

// ------------------------- host launcher -----------------------------------
extern "C" void kernel_launch(void* const* d_in, const int* in_sizes, int n_in,
                              void* d_out, int out_size, void* d_ws, size_t ws_size,
                              hipStream_t stream) {
  (void)in_sizes; (void)n_in; (void)out_size; (void)ws_size;
  auto F = [&](int i) { return (const float*)d_in[i]; };
  auto I = [&](int i) { return (const int*)d_in[i]; };

  const int S_ = 128, V_ = 16;
  const int NPH = 4096, NPR = 16384;

  float* ws       = (float*)d_ws;
  float* cur_s_ph = ws;
  float* cur_v_ph = cur_s_ph + NPH * S_;
  float* cur_s_pr = cur_v_ph + NPH * V_ * 3;
  float* cur_v_pr = cur_s_pr + NPR * S_;
  float* agg_s_ph = cur_v_pr + NPR * V_ * 3;
  float* agg_v_ph = agg_s_ph + NPH * S_;
  float* agg_s_pr = agg_v_ph + NPH * V_ * 3;
  float* agg_v_pr = agg_s_pr + NPR * S_;
  const int aggN  = NPH * S_ + NPH * V_ * 3 + NPR * S_ + NPR * V_ * 3;

  // packed fragment pool (32B aligned: node+agg region is a multiple of 8 floats)
  unsigned short* pk_us = (unsigned short*)(agg_v_pr + NPR * V_ * 3);
  const v16bf*    pk    = (const v16bf*)pk_us;
  int fragOff = 0;

  copy_kernel<<<(NPH * S_ + 255) / 256, 256, 0, stream>>>(cur_s_ph, F(0), NPH * S_);
  copy_kernel<<<(NPH * 48 + 255) / 256, 256, 0, stream>>>(cur_v_ph, F(2), NPH * 48);
  copy_kernel<<<(NPR * S_ + 255) / 256, 256, 0, stream>>>(cur_s_pr, F(3), NPR * S_);
  copy_kernel<<<(NPR * 48 + 255) / 256, 256, 0, stream>>>(cur_v_pr, F(5), NPR * 48);

  // pack one weight matrix -> bf16 fragments; returns offset in v16bf units
  auto packB = [&](int srcIdx, int K, int N, int KPAD, int NPAD) -> int {
    const int KT = KPAD / 32, NT = NPAD / 16;
    const int frags = NT * KT * 32;
    unsigned short* dst = pk_us + (size_t)fragOff * 16;
    pack_B_kernel<<<(frags * 16 + 255) / 256, 256, 0, stream>>>(F(srcIdx), K, N,
                                                                KT, NT, dst);
    const int r = fragOff;
    fragOff += frags;
    return r;
  };
  // params pytree flatten order (sorted dict keys): 14 W_out, 15 b_out,
  // 16..399 convs (ets ff,fp,pf,pp; upd pharm,prot; per-GVP W,Wg,Wh,Wu,b,bg),
  // 400..417 noise.
  auto mko = [&](int base, int vin, int vout, int sin, int sout) {
    const int h = vin > vout ? vin : vout;
    PackedOff o;
    o.W  = packB(base + 0, sin + h, sout, ((sin + h + 31) / 32) * 32, sout);
    o.Wg = packB(base + 1, sout, vout, sout, 16);
    o.Wh = packB(base + 2, vin, h, 32, 32);
    o.Wu = packB(base + 3, h, vout, 32, 16);
    return o;
  };

  const int oWout = packB(14, 64, 128, 64, 128);

  for (int l = 0; l < 4; ++l) {
    fill_kernel<<<2048, 256, 0, stream>>>(agg_s_ph, 0.f, aggN);
    const int base  = 16 + 96 * l;
    const int ubase = base + 72;

    // ff (et 0): pharm -> pharm
    {
      PackedOff o0 = mko(base + 0, 17, 16, 129, 128);
      PackedOff o1 = mko(base + 6, 16, 16, 128, 128);
      PackedOff o2 = mko(base + 12, 16, 16, 128, 128);
      gvp_msg_kernel<<<16384 / 16, 32, 0, stream>>>(
          cur_s_ph, cur_v_ph, F(1), F(1), I(6), I(7), 16384, pk, o0, o1, o2,
          F(base + 4), F(base + 5), F(base + 10), F(base + 11), F(base + 16),
          F(base + 17), agg_s_ph, agg_v_ph);
    }
    // fp (et 1): pharm -> prot
    {
      PackedOff o0 = mko(base + 18, 17, 16, 129, 128);
      PackedOff o1 = mko(base + 24, 16, 16, 128, 128);
      PackedOff o2 = mko(base + 30, 16, 16, 128, 128);
      gvp_msg_kernel<<<16384 / 16, 32, 0, stream>>>(
          cur_s_ph, cur_v_ph, F(1), F(4), I(10), I(11), 16384, pk, o0, o1, o2,
          F(base + 22), F(base + 23), F(base + 28), F(base + 29), F(base + 34),
          F(base + 35), agg_s_pr, agg_v_pr);
    }
    // pf (et 2): prot -> pharm
    {
      PackedOff o0 = mko(base + 36, 17, 16, 129, 128);
      PackedOff o1 = mko(base + 42, 16, 16, 128, 128);
      PackedOff o2 = mko(base + 48, 16, 16, 128, 128);
      gvp_msg_kernel<<<16384 / 16, 32, 0, stream>>>(
          cur_s_pr, cur_v_pr, F(4), F(1), I(8), I(9), 16384, pk, o0, o1, o2,
          F(base + 40), F(base + 41), F(base + 46), F(base + 47), F(base + 52),
          F(base + 53), agg_s_ph, agg_v_ph);
    }
    // pp (et 3): prot -> prot
    {
      PackedOff o0 = mko(base + 54, 17, 16, 129, 128);
      PackedOff o1 = mko(base + 60, 16, 16, 128, 128);
      PackedOff o2 = mko(base + 66, 16, 16, 128, 128);
      gvp_msg_kernel<<<32768 / 16, 32, 0, stream>>>(
          cur_s_pr, cur_v_pr, F(4), F(4), I(12), I(13), 32768, pk, o0, o1, o2,
          F(base + 58), F(base + 59), F(base + 64), F(base + 65), F(base + 70),
          F(base + 71), agg_s_pr, agg_v_pr);
    }
    // updates: pharm (nt 0), prot (nt 1)
    {
      PackedOff o0 = mko(ubase + 0, 32, 16, 256, 128);
      PackedOff o1 = mko(ubase + 6, 16, 16, 128, 128);
      gvp_upd_kernel<<<NPH / 16, 32, 0, stream>>>(
          cur_s_ph, cur_v_ph, agg_s_ph, agg_v_ph, NPH, pk, o0, o1,
          F(ubase + 4), F(ubase + 5), F(ubase + 10), F(ubase + 11));
    }
    {
      PackedOff o0 = mko(ubase + 12, 32, 16, 256, 128);
      PackedOff o1 = mko(ubase + 18, 16, 16, 128, 128);
      gvp_upd_kernel<<<NPR / 16, 32, 0, stream>>>(
          cur_s_pr, cur_v_pr, agg_s_pr, agg_v_pr, NPR, pk, o0, o1,
          F(ubase + 16), F(ubase + 17), F(ubase + 22), F(ubase + 23));
    }
  }

  const int nbase = 16 + 384;
  PackedOff no0 = mko(nbase + 0, 16, 16, 128, 128);
  PackedOff no1 = mko(nbase + 6, 16, 16, 128, 128);
  PackedOff no2 = mko(nbase + 12, 16, 1, 128, 64);
  float* out = (float*)d_out;
  gvp_noise_kernel<<<NPH / 16, 32, 0, stream>>>(
      cur_s_ph, cur_v_ph, NPH, pk, no0, no1, no2, oWout,
      F(nbase + 4), F(nbase + 5), F(nbase + 10), F(nbase + 11), F(nbase + 16),
      F(nbase + 17), F(15), out, out + NPH * S_);
}
